// ImageBoxSampling_29025388987084
// MI455X (gfx1250) — matrix operand order repaired and checked by reading, not compiled, not run on hardware
//
#include <hip/hip_runtime.h>
#include <hip/hip_bf16.h>

#define IMG_H 448
#define IMG_W 800
#define POOL  16
#define PG    14            // 224/16 pooled grid
#define NCAM  6
#define NBOX  48
#define NCLS  10
#define KDIM  588
#define KPAD  608           // 19*32, zero-padded K for WMMA
#define KB32  (KPAD/32)     // 19 K-blocks
#define DDIM  768
#define MROWS (NCAM*NBOX)   // 288 = 18 tiles of 16
#define TM_T  (MROWS/16)    // 18
#define TN_T  (DDIM/16)     // 48
#define NT    4             // tn tiles per wave in GEMM

typedef __attribute__((ext_vector_type(16))) _Float16 v16h;
typedef __attribute__((ext_vector_type(8)))  float    v8f;

// ---------------------------------------------------------------------------
// Kernel 1: corners -> projection -> bbox {x1, y1, sq, visible} per (cam,box)
// ---------------------------------------------------------------------------
__global__ void k_bbox(const float* __restrict__ boxes7,
                       const float* __restrict__ l2i,
                       const float* __restrict__ img_aug,
                       const float* __restrict__ lidar_aug,
                       float4* __restrict__ bbox) {
  int idx = blockIdx.x * blockDim.x + threadIdx.x;
  if (idx >= NCAM * NBOX) return;
  int cam = idx / NBOX, n = idx % NBOX;
  const float* b = boxes7 + n * 7;
  float cx = b[0], cy = b[1], cz = b[2];
  float dx = b[3] * 0.5f, dy = b[4] * 0.5f, dz = b[5] * 0.5f;
  float ch = __cosf(b[6]), sh = __sinf(b[6]);

  // lidar_aug: pts = inv(R) @ (corners - t)
  float L[9];
  #pragma unroll
  for (int i = 0; i < 3; ++i)
    #pragma unroll
    for (int j = 0; j < 3; ++j) L[i * 3 + j] = lidar_aug[i * 4 + j];
  float tx = lidar_aug[3], ty = lidar_aug[7], tz = lidar_aug[11];
  float det = L[0]*(L[4]*L[8]-L[5]*L[7]) - L[1]*(L[3]*L[8]-L[5]*L[6])
            + L[2]*(L[3]*L[7]-L[4]*L[6]);
  float id_ = 1.0f / det;
  float inv[9] = {
    (L[4]*L[8]-L[5]*L[7])*id_, (L[2]*L[7]-L[1]*L[8])*id_, (L[1]*L[5]-L[2]*L[4])*id_,
    (L[5]*L[6]-L[3]*L[8])*id_, (L[0]*L[8]-L[2]*L[6])*id_, (L[2]*L[3]-L[0]*L[5])*id_,
    (L[3]*L[7]-L[4]*L[6])*id_, (L[1]*L[6]-L[0]*L[7])*id_, (L[0]*L[4]-L[1]*L[3])*id_ };

  const float* M = l2i + cam * 16;     // rows 0..2 of 4x4
  const float* A = img_aug + cam * 16;
  const float tmpl[8][3] = {{1,1,-1},{1,-1,-1},{-1,-1,-1},{-1,1,-1},
                            {1,1,1},{1,-1,1},{-1,-1,1},{-1,1,1}};
  float minr = 3e9f, maxr = -3e9f, minc = 3e9f, maxc = -3e9f;
  int vis = 0;
  #pragma unroll
  for (int k = 0; k < 8; ++k) {
    float lx = tmpl[k][0]*dx, ly = tmpl[k][1]*dy, lz = tmpl[k][2]*dz;
    // rotated[j] = sum_i local[i]*R[i][j], R = [[c,s,0],[-s,c,0],[0,0,1]]
    float wx = lx*ch - ly*sh + cx;
    float wy = lx*sh + ly*ch + cy;
    float wz = lz + cz;
    float px = wx - tx, py = wy - ty, pz = wz - tz;
    float qx = inv[0]*px + inv[1]*py + inv[2]*pz;
    float qy = inv[3]*px + inv[4]*py + inv[5]*pz;
    float qz = inv[6]*px + inv[7]*py + inv[8]*pz;
    float ux = M[0]*qx + M[1]*qy + M[2]*qz  + M[3];
    float uy = M[4]*qx + M[5]*qy + M[6]*qz  + M[7];
    float uz = M[8]*qx + M[9]*qy + M[10]*qz + M[11];
    float zc = fminf(fmaxf(uz, 1e-5f), 1e5f);
    float sx = ux / zc, sy = uy / zc;
    float ax = A[0]*sx + A[1]*sy + A[2]*zc + A[3];
    float ay = A[4]*sx + A[5]*sy + A[6]*zc + A[7];
    float row = ay, col = ax;
    vis |= (row < (float)IMG_H) && (row >= 0.f) && (col < (float)IMG_W) && (col >= 0.f);
    float tr = truncf(fminf(fmaxf(row, -2e9f), 2e9f));
    float tc = truncf(fminf(fmaxf(col, -2e9f), 2e9f));
    minr = fminf(minr, tr); maxr = fmaxf(maxr, tr);
    minc = fminf(minc, tc); maxc = fmaxf(maxc, tc);
  }
  float x1 = fminf(fmaxf(minc, 0.f), (float)IMG_W);
  float x2 = fminf(fmaxf(maxc, 0.f), (float)IMG_W);
  float y1 = fminf(fmaxf(minr, 0.f), (float)IMG_H);
  float y2 = fminf(fmaxf(maxr, 0.f), (float)IMG_H);
  float sq = fmaxf(fmaxf(x2 - x1, y2 - y1), 64.0f);
  bbox[idx] = make_float4(x1, y1, sq, vis ? 1.0f : 0.0f);
}

// ---------------------------------------------------------------------------
// Kernel 2: fused bilinear sample (224x224) + 16x16 avg pool -> A_f16 (288xKPAD)
// One block per (cam,box,channel); one pooled cell per thread (196 active).
// ---------------------------------------------------------------------------
__global__ void k_sample_pool(const float* __restrict__ imgs,
                              const float4* __restrict__ bbox,
                              _Float16* __restrict__ Aout) {
  int blk = blockIdx.x;          // cam*NBOX*3 + box*3 + ch
  int ch  = blk % 3;
  int cb  = blk / 3;             // cam*NBOX + box
  int tid = threadIdx.x;
  // zero the K padding columns once per row (ch==0 block)
  if (ch == 0 && tid >= PG*PG && tid < PG*PG + (KPAD - KDIM))
    Aout[(size_t)cb * KPAD + KDIM + (tid - PG*PG)] = (_Float16)0.0f;
  if (tid >= PG * PG) return;
  int pr = tid / PG, pc = tid % PG;
  int cam = cb / NBOX;
  float4 bb = bbox[cb];
  const float* img = imgs + ((size_t)(cam * 3 + ch)) * IMG_H * IMG_W;
  const float inv223 = 1.0f / 223.0f;
  float acc = 0.0f;
  for (int jy = 0; jy < POOL; ++jy) {
    int gy = pr * POOL + jy;
    float iy  = bb.y + bb.z * ((float)gy * inv223) - 0.5f;   // unnormalized sample y
    float y0f = floorf(iy);
    float wy1 = iy - y0f, wy0 = 1.0f - wy1;
    int  y0  = (int)y0f;
    bool vy0 = (y0f >= 0.f) && (y0f <= (float)(IMG_H - 1));
    bool vy1 = (y0f + 1.f >= 0.f) && (y0f + 1.f <= (float)(IMG_H - 1));
    int y0c = min(max(y0, 0),     IMG_H - 1);
    int y1c = min(max(y0 + 1, 0), IMG_H - 1);
    for (int jx = 0; jx < POOL; ++jx) {
      int gx = pc * POOL + jx;
      float ix  = bb.x + bb.z * ((float)gx * inv223) - 0.5f;
      float x0f = floorf(ix);
      float wx1 = ix - x0f, wx0 = 1.0f - wx1;
      int  x0  = (int)x0f;
      bool vx0 = (x0f >= 0.f) && (x0f <= (float)(IMG_W - 1));
      bool vx1 = (x0f + 1.f >= 0.f) && (x0f + 1.f <= (float)(IMG_W - 1));
      int x0c = min(max(x0, 0),     IMG_W - 1);
      int x1c = min(max(x0 + 1, 0), IMG_W - 1);
      float v00 = (vx0 && vy0) ? img[y0c * IMG_W + x0c] : 0.f;
      float v10 = (vx1 && vy0) ? img[y0c * IMG_W + x1c] : 0.f;
      float v01 = (vx0 && vy1) ? img[y1c * IMG_W + x0c] : 0.f;
      float v11 = (vx1 && vy1) ? img[y1c * IMG_W + x1c] : 0.f;
      acc += v00 * wx0 * wy0 + v10 * wx1 * wy0 + v01 * wx0 * wy1 + v11 * wx1 * wy1;
    }
  }
  Aout[(size_t)cb * KPAD + ch * (PG*PG) + pr * PG + pc] = (_Float16)(acc * (1.0f / 256.0f));
}

// ---------------------------------------------------------------------------
// Kernel 3: cast + swizzle W_img (588x768) into WMMA-fragment-ordered f16 B:
//   Bs[((tn*KB32 + kb)*32 + lane)*16 + t] = W[kb*32 + (lane>>4)*16 + t][tn*16 + (lane&15)]
// Each fragment becomes one contiguous, 32B-aligned 16-half record so the GEMM
// reads it with a single v16h (2x global_load_b128) load.
// ---------------------------------------------------------------------------
__global__ void k_wswizzle(const float* __restrict__ W, _Float16* __restrict__ Bs) {
  int frag = blockIdx.x * blockDim.x + threadIdx.x;
  const int NFRAG = TN_T * KB32 * 32;        // 48*19*32 = 29184
  if (frag >= NFRAG) return;
  int lane = frag & 31;
  int kb   = (frag >> 5) % KB32;
  int tn   = (frag >> 5) / KB32;
  int lh = lane & 15, hi = lane >> 4;
  int col   = tn * 16 + lh;
  int kbase = kb * 32 + hi * 16;
  _Float16* dst = Bs + (size_t)frag * 16;
  #pragma unroll
  for (int t = 0; t < 16; ++t) {
    int r = kbase + t;
    dst[t] = (r < KDIM) ? (_Float16)W[(size_t)r * DDIM + col] : (_Float16)0.0f;
  }
}

// ---------------------------------------------------------------------------
// Kernel 4: C(288x768) = A(288xKPAD) x B(KPADx768) via v_wmma_f32_16x16x32_f16
// One wave handles one tm tile x NT=4 tn tiles: A fragment loaded once per
// K-step, 4 independent WMMAs issued back-to-back (XDL-friendly), B fragments
// read from the pre-swizzled contiguous layout.
// ---------------------------------------------------------------------------
__global__ void k_gemm_wmma(const _Float16* __restrict__ Af,
                            const _Float16* __restrict__ Bs,
                            float* __restrict__ C) {
  int wave = threadIdx.x >> 5;
  int lane = threadIdx.x & 31;
  int wid  = blockIdx.x * (blockDim.x >> 5) + wave;   // 0 .. TM_T*(TN_T/NT)-1
  if (wid >= TM_T * (TN_T / NT)) return;              // wave-uniform guard
  int tm  = wid % TM_T;
  int tn0 = (wid / TM_T) * NT;
  int lh = lane & 15, hi = lane >> 4;
  int m  = tm * 16 + lh;                              // A row for this lane
  v8f c[NT] = {};
  const _Float16* arow = Af + (size_t)m * KPAD + hi * 16;
  for (int kb = 0; kb < KB32; ++kb) {
    v16h a;
    // A 16x32 f16 layout: lane-half hi selects K {0..15} vs {16..31}
    const _Float16* pa = arow + kb * 32;
    #pragma unroll
    for (int t = 0; t < 8; ++t) { a[t] = pa[t]; a[8 + t] = pa[8 + t]; }
    #pragma unroll
    for (int j = 0; j < NT; ++j) {
      const v16h* pb = (const v16h*)(Bs +
          ((size_t)((tn0 + j) * KB32 + kb) * 32 + lane) * 16);
      v16h bfrag = *pb;
      c[j] = __builtin_amdgcn_wmma_f32_16x16x32_f16(
          /*neg_a=*/false, a, /*neg_b=*/false, bfrag,
          /*c_mod=*/(short)0, c[j], /*reuse_a=*/false, /*reuse_b=*/false);
    }
  }
  #pragma unroll
  for (int j = 0; j < NT; ++j)
    #pragma unroll
    for (int i = 0; i < 8; ++i)
      C[(size_t)(tm * 16 + hi * 8 + i) * DDIM + (tn0 + j) * 16 + lh] = c[j][i];
}

// ---------------------------------------------------------------------------
// Kernel 5: normalize feats, cosine logits vs text, softmax, visibility-masked
// camera mean, argmax. One block per box; LDS tree reductions.
// out layout: pred[0..48) true[48..96) acc[96] mean_probs[97..577)
// ---------------------------------------------------------------------------
__global__ void k_probs(const float* __restrict__ C,
                        const float* __restrict__ text,
                        const float* __restrict__ logit_scale,
                        const int* __restrict__ labels,
                        const float4* __restrict__ bbox,
                        float* __restrict__ out) {
  __shared__ float red[256];
  __shared__ float tinv[NCLS];
  __shared__ float slog[NCLS];
  int n = blockIdx.x;
  int tid = threadIdx.x;
  // text inverse norms
  for (int k = 0; k < NCLS; ++k) {
    float s = 0.f;
    for (int d = tid; d < DDIM; d += 256) { float t = text[k * DDIM + d]; s += t * t; }
    red[tid] = s; __syncthreads();
    for (int o = 128; o > 0; o >>= 1) { if (tid < o) red[tid] += red[tid + o]; __syncthreads(); }
    if (tid == 0) tinv[k] = rsqrtf(red[0]);
    __syncthreads();
  }
  float scale = __expf(logit_scale[0]);
  float psum[NCLS];
  #pragma unroll
  for (int k = 0; k < NCLS; ++k) psum[k] = 0.f;
  float cnt = 0.f;
  for (int cam = 0; cam < NCAM; ++cam) {
    int row = cam * NBOX + n;
    const float* f = C + (size_t)row * DDIM;
    // feature row inverse norm
    float s = 0.f;
    for (int d = tid; d < DDIM; d += 256) { float v = f[d]; s += v * v; }
    red[tid] = s; __syncthreads();
    for (int o = 128; o > 0; o >>= 1) { if (tid < o) red[tid] += red[tid + o]; __syncthreads(); }
    float finv = rsqrtf(fmaxf(red[0], 1e-30f));
    __syncthreads();
    // 10 dot products
    float loc[NCLS];
    #pragma unroll
    for (int k = 0; k < NCLS; ++k) loc[k] = 0.f;
    for (int d = tid; d < DDIM; d += 256) {
      float v = f[d];
      #pragma unroll
      for (int k = 0; k < NCLS; ++k) loc[k] += v * text[k * DDIM + d];
    }
    for (int k = 0; k < NCLS; ++k) {
      red[tid] = loc[k]; __syncthreads();
      for (int o = 128; o > 0; o >>= 1) { if (tid < o) red[tid] += red[tid + o]; __syncthreads(); }
      if (tid == 0) slog[k] = scale * red[0] * finv * tinv[k];
      __syncthreads();
    }
    if (tid == 0) {
      float vis = bbox[row].w;
      float mx = slog[0];
      #pragma unroll
      for (int k = 1; k < NCLS; ++k) mx = fmaxf(mx, slog[k]);
      float se = 0.f, e[NCLS];
      #pragma unroll
      for (int k = 0; k < NCLS; ++k) { e[k] = __expf(slog[k] - mx); se += e[k]; }
      #pragma unroll
      for (int k = 0; k < NCLS; ++k) psum[k] += (e[k] / se) * vis;
      cnt += vis;
    }
    __syncthreads();
  }
  if (tid == 0) {
    float icnt = 1.0f / fmaxf(cnt, 1.0f);
    int best = 0; float bv = -1.f;
    #pragma unroll
    for (int k = 0; k < NCLS; ++k) {
      float mp = psum[k] * icnt;
      out[97 + n * NCLS + k] = mp;
      if (mp > bv) { bv = mp; best = k; }    // first-max wins, matches jnp.argmax
    }
    out[n] = (float)best;
    out[NBOX + n] = (float)(labels[n] - 1);
  }
}

// ---------------------------------------------------------------------------
// Kernel 6: accuracy over 48 boxes
// ---------------------------------------------------------------------------
__global__ void k_acc(float* __restrict__ out) {
  __shared__ float red[64];
  int tid = threadIdx.x;
  float v = 0.f;
  if (tid < NBOX) v = (out[tid] == out[NBOX + tid]) ? 1.0f : 0.0f;
  red[tid] = v; __syncthreads();
  for (int o = 32; o > 0; o >>= 1) { if (tid < o) red[tid] += red[tid + o]; __syncthreads(); }
  if (tid == 0) out[96] = red[0] * (1.0f / (float)NBOX);
}

extern "C" void kernel_launch(void* const* d_in, const int* in_sizes, int n_in,
                              void* d_out, int out_size, void* d_ws, size_t ws_size,
                              hipStream_t stream) {
  const float* boxes7 = (const float*)d_in[0];
  const int*   labels = (const int*)d_in[1];
  const float* imgs   = (const float*)d_in[2];
  const float* l2i    = (const float*)d_in[3];
  const float* iaug   = (const float*)d_in[4];
  const float* laug   = (const float*)d_in[5];
  const float* Wimg   = (const float*)d_in[6];
  const float* text   = (const float*)d_in[7];
  const float* lscale = (const float*)d_in[8];
  (void)in_sizes; (void)n_in; (void)out_size; (void)ws_size;

  char* ws = (char*)d_ws;
  float4*   bbox = (float4*)ws;                                    //   4,608 B
  _Float16* Af   = (_Float16*)(ws + 4608);                         // 350,208 B
  _Float16* Bs   = (_Float16*)(ws + 4608 + 350208);                // 933,888 B (32B-aligned)
  float*    C    = (float*)(ws + 4608 + 350208 + 933888);          // 884,736 B
  float*    out  = (float*)d_out;

  k_bbox<<<1, MROWS, 0, stream>>>(boxes7, l2i, iaug, laug, bbox);
  k_wswizzle<<<(TN_T * KB32 * 32 + 255) / 256, 256, 0, stream>>>(Wimg, Bs);
  k_sample_pool<<<NCAM * NBOX * 3, 256, 0, stream>>>(imgs, bbox, Af);
  k_gemm_wmma<<<(TM_T * (TN_T / NT) + 3) / 4, 128, 0, stream>>>(Af, Bs, C);
  k_probs<<<NBOX, 256, 0, stream>>>(C, text, lscale, labels, bbox, out);
  k_acc<<<1, 64, 0, stream>>>(out);
}
